// FloquetRecurrentSolver_90701119357170
// MI455X (gfx1250) — matrix operand order repaired and checked by reading, not compiled, not run on hardware
//
#include <hip/hip_runtime.h>
#include <hip/hip_bf16.h>
#include <cstdint>
#include <cstddef>

// ---------------------------------------------------------------------------
// Problem constants (fixed by the harness / reference setup)
// ---------------------------------------------------------------------------
constexpr int GB    = 256;     // graphs (B)
constexpr int NNPG  = 36;      // nodes per graph (dimq*(2*bz+1))
constexpr int NTOT  = 9216;    // total nodes
constexpr int NE    = 65536;   // edges
constexpr int DIMQ  = 4;
constexpr int BZ    = 4;
constexpr int ROOT0 = DIMQ * BZ; // 16

typedef __attribute__((ext_vector_type(16))) _Float16 v16h;
typedef __attribute__((ext_vector_type(8)))  _Float16 v8h;
typedef __attribute__((ext_vector_type(8)))  float    v8f;

// ---------------------------------------------------------------------------
// Workspace layout (bytes)
// ---------------------------------------------------------------------------
constexpr size_t FR_ENC2      = 0;                       // halfs; 2 ktiles
constexpr size_t FR_LAYER(int l) { return 4096 + (size_t)l * 30720; }
// per-layer fragment sub-offsets (halfs): w1_0 +0 (3kt), w1_1 +6144, w1_2 +10240,
// w2_0 +14336 (4kt), w2_1 +22528, w2_2 +26624
constexpr size_t SZ_WFRAG_H   = 4096 + 5 * 30720;        // 157,696 halfs
constexpr size_t OFF_WFRAG    = 0;
constexpr size_t OFF_OFFS     = OFF_WFRAG + SZ_WFRAG_H * 2;          // 315,392
constexpr size_t OFF_H        = OFF_OFFS  + 1024 * 4;                // h     f16 [NTOT*64]
constexpr size_t OFF_T1       = OFF_H     + (size_t)NTOT * 64 * 2;   // t1    f16 [NTOT*64]
constexpr size_t OFF_T2       = OFF_T1    + (size_t)NTOT * 64 * 2;   // t2    f16 [NTOT*64]
constexpr size_t OFF_XNODE    = OFF_T2    + (size_t)NTOT * 64 * 2;   // xnode f16 [NTOT*128]
constexpr size_t OFF_AGG      = OFF_XNODE + (size_t)NTOT * 128 * 2;  // agg   f32 [NTOT*64]
constexpr size_t OFF_XEDGE    = OFF_AGG   + (size_t)NTOT * 64 * 4;   // xedge f16 [NE*96]
constexpr size_t OFF_MA       = OFF_XEDGE + (size_t)NE * 96 * 2;     // mA    f16 [NE*64]
constexpr size_t OFF_MB       = OFF_MA    + (size_t)NE * 64 * 2;     // mB    f16 [NE*64]

// ---------------------------------------------------------------------------
// Weight repack: f32 [K x 64] row-major -> f16 WMMA B-fragment layout.
// Fragment element ((kt*4+nt)*32+lane)*16 + v*2 + h holds
//   W[kt*32 + ((lane>=16)?16:0) + 2v + h][nt*16 + (lane&15)]   (0 if k>=Kreal)
// ---------------------------------------------------------------------------
__global__ void fl_wconv(const float* __restrict__ W, _Float16* __restrict__ frag,
                         int kreal, int ktiles) {
  int idx = blockIdx.x * blockDim.x + threadIdx.x;
  int total = ktiles * 2048;
  if (idx >= total) return;
  int h    = idx & 1;
  int v    = (idx >> 1) & 7;
  int lane = (idx >> 4) & 31;
  int nt   = (idx >> 9) & 3;
  int kt   = idx >> 11;
  int kbase = (lane >= 16) ? 16 : 0;
  int k = kt * 32 + kbase + 2 * v + h;
  int n = nt * 16 + (lane & 15);
  float val = (k < kreal) ? W[(size_t)k * 64 + n] : 0.f;
  frag[idx] = (_Float16)val;
}

// ---------------------------------------------------------------------------
// Core WMMA GEMM:  Y[M x 64] = act(X[M x Kpad](f16) @ Wfrag + bias)
// Each wave owns MT consecutive 16-row tiles x 64 cols (MT*4 v8f accumulators);
// B fragments are loaded once per (kt,nt) and reused across the MT A fragments,
// raising the wmma:vmem issue ratio (16 loads : 16 wmma at MT=4 vs 10:4 at MT=1).
// SCATTER epilogue: atomic f32 add into agg[dst[row]*64 + col] (fused segment_sum).
// ---------------------------------------------------------------------------
template <int KTILES, int MT, bool RELU, bool SCATTER>
__global__ void fl_gemm(const _Float16* __restrict__ X, int ldx,
                        const _Float16* __restrict__ frag,
                        const float* __restrict__ bias,
                        _Float16* __restrict__ Yh,
                        float* __restrict__ aggOut,
                        const long long* __restrict__ dstIdx) {
  const int lane  = threadIdx.x & 31;
  const int wave  = threadIdx.x >> 5;
  const int tile0 = (blockIdx.x * 8 + wave) * MT;
  const int row0  = tile0 * 16;
  const int r     = lane & 15;
  const int koff  = (lane & 16) ? 16 : 0;
  const _Float16* xrow = X + (size_t)(row0 + r) * ldx + koff;

  v8f acc[MT][4] = {};
#pragma unroll
  for (int kt = 0; kt < KTILES; ++kt) {
    // A fragments: lane<16 -> K {0..15}, lane>=16 -> K {16..31}
    v16h a[MT];
#pragma unroll
    for (int m = 0; m < MT; ++m) {
      const _Float16* p = xrow + (size_t)m * 16 * ldx + kt * 32;
      const v8h alo = *(const v8h*)(p);
      const v8h ahi = *(const v8h*)(p + 8);
      a[m] = __builtin_shufflevector(alo, ahi, 0, 1, 2, 3, 4, 5, 6, 7,
                                     8, 9, 10, 11, 12, 13, 14, 15);
    }
#pragma unroll
    for (int nt = 0; nt < 4; ++nt) {
      const v16h b = *(const v16h*)(frag + ((size_t)(kt * 4 + nt) * 32 + lane) * 16);
#pragma unroll
      for (int m = 0; m < MT; ++m) {
        acc[m][nt] = __builtin_amdgcn_wmma_f32_16x16x32_f16(
            false, a[m], false, b, (short)0, acc[m][nt], false, false);
      }
    }
  }

  const int colbase = lane & 15;
  const int rsel    = (lane & 16) ? 8 : 0;
#pragma unroll
  for (int m = 0; m < MT; ++m) {
#pragma unroll
    for (int nt = 0; nt < 4; ++nt) {
      const int col = nt * 16 + colbase;
      const float bv = bias[col];
#pragma unroll
      for (int j = 0; j < 8; ++j) {
        const int row = row0 + m * 16 + j + rsel;
        float v = acc[m][nt][j] + bv;
        if (RELU) v = v > 0.f ? v : 0.f;
        if (SCATTER) {
          const long long d = dstIdx[row];
          unsafeAtomicAdd(aggOut + (size_t)d * 64 + col, v);
        } else {
          Yh[(size_t)row * 64 + col] = (_Float16)v;
        }
      }
    }
  }
}

// ---------------------------------------------------------------------------
// Small helper kernels
// ---------------------------------------------------------------------------
__global__ void fl_offsets(const float* __restrict__ x, float* __restrict__ offs) {
  int idx = blockIdx.x * blockDim.x + threadIdx.x;  // 1024
  if (idx >= 4 * GB) return;
  int i = idx >> 8, b = idx & 255;
  int root = b * NNPG + ROOT0 + i;
  offs[i * GB + b] = x[root * 3 + 0];
}

__global__ void fl_zero_f32(float* __restrict__ p, int n) {
  int i = blockIdx.x * blockDim.x + threadIdx.x;
  if (i < n) p[i] = 0.f;
}

// Encoder layer 1: xi (built inline) @ enc_w1 + b1, relu -> f16 [NTOT x 64]
__global__ void fl_encode(const float* __restrict__ x, const float* __restrict__ offs,
                          int i, const float* __restrict__ w1,
                          const float* __restrict__ b1, _Float16* __restrict__ out) {
  int idx = blockIdx.x * blockDim.x + threadIdx.x;  // NTOT*64
  if (idx >= NTOT * 64) return;
  int n = idx >> 6, f = idx & 63;
  int b = n / NNPG;
  float off = offs[b];
  float x0 = x[n * 3 + 0], x1 = x[n * 3 + 1], x2 = x[n * 3 + 2];
  float xi0 = x0 - off;
  bool cond = (xi0 == 0.f) && ((n & 3) == i);
  float col3 = cond ? 1.f : x2;
  float acc = b1[f] + xi0 * w1[0 * 64 + f] + x0 * w1[1 * 64 + f] +
              x1 * w1[2 * 64 + f] + col3 * w1[3 * 64 + f];
  out[(size_t)n * 64 + f] = (_Float16)(acc > 0.f ? acc : 0.f);
}

// Build edge GEMM input rows: [h[src] (64) | edge_attr (8) | zero pad (24)]
__global__ void fl_gather_edge(const _Float16* __restrict__ h,
                               const float* __restrict__ eattr,
                               const long long* __restrict__ srcIdx,
                               _Float16* __restrict__ Xe) {
  int idx = blockIdx.x * blockDim.x + threadIdx.x;  // NE*96
  if (idx >= NE * 96) return;
  int e = idx / 96, c = idx % 96;
  long long s = srcIdx[e];
  _Float16 v;
  if (c < 64)       v = h[(size_t)s * 64 + c];
  else if (c < 72)  v = (_Float16)eattr[(size_t)e * 8 + (c - 64)];
  else              v = (_Float16)0.f;
  Xe[(size_t)e * 96 + c] = v;
}

// Build node GEMM input rows: [h (64) | agg (64)]
__global__ void fl_xnode(const _Float16* __restrict__ h, const float* __restrict__ agg,
                         _Float16* __restrict__ Xn) {
  int idx = blockIdx.x * blockDim.x + threadIdx.x;  // NTOT*128
  if (idx >= NTOT * 128) return;
  int n = idx >> 7, c = idx & 127;
  Xn[(size_t)n * 128 + c] =
      (c < 64) ? h[(size_t)n * 64 + c] : (_Float16)agg[(size_t)n * 64 + (c - 64)];
}

// Floquet collapse (sum over 9 copies) + dur MLP (64->64 relu ->1); fills row/col i.
__global__ void fl_dur(const _Float16* __restrict__ h,
                       const float* __restrict__ w1, const float* __restrict__ b1,
                       const float* __restrict__ w2, const float* __restrict__ b2,
                       float* __restrict__ out, int i) {
  __shared__ float enc[64];
  __shared__ float hid[64];
  int bq = blockIdx.x;          // 1024 blocks
  int b = bq >> 2, q = bq & 3;
  int t = threadIdx.x;          // 64 threads
  float s = 0.f;
  for (int c = 0; c < 9; ++c)
    s += (float)h[((size_t)(b * NNPG + c * DIMQ + q)) * 64 + t];
  enc[t] = s;
  __syncthreads();
  float a = b1[t];
  for (int k = 0; k < 64; ++k) a += enc[k] * w1[k * 64 + t];
  hid[t] = a > 0.f ? a : 0.f;
  __syncthreads();
  if (t == 0) {
    float d = b2[0];
    for (int k = 0; k < 64; ++k) d += hid[k] * w2[k];
    out[b * 16 + i * 4 + q] = d;   // matrix[:, i, :]
    out[b * 16 + q * 4 + i] = d;   // matrix[:, :, i]
  }
}

// Diagonal decoder: [root_feat(64) | offset] -> 256 relu -> 64 relu -> 1
__global__ void fl_dec(const _Float16* __restrict__ h, const float* __restrict__ offs,
                       int i,
                       const float* __restrict__ w1, const float* __restrict__ b1,
                       const float* __restrict__ w2, const float* __restrict__ b2,
                       const float* __restrict__ w3, const float* __restrict__ b3,
                       float* __restrict__ out) {
  __shared__ float in65[65];
  __shared__ float a1[256];
  __shared__ float a2[64];
  int b = blockIdx.x;          // 256 blocks
  int t = threadIdx.x;         // 256 threads
  size_t root = (size_t)b * NNPG + ROOT0 + i;
  if (t < 64) in65[t] = (float)h[root * 64 + t];
  if (t == 0) in65[64] = offs[b];
  __syncthreads();
  float a = b1[t];
  for (int k = 0; k < 65; ++k) a += in65[k] * w1[k * 256 + t];
  a1[t] = a > 0.f ? a : 0.f;
  __syncthreads();
  if (t < 64) {
    float v = b2[t];
    for (int k = 0; k < 256; ++k) v += a1[k] * w2[k * 64 + t];
    a2[t] = v > 0.f ? v : 0.f;
  }
  __syncthreads();
  if (t == 0) {
    float d = b3[0];
    for (int k = 0; k < 64; ++k) d += a2[k] * w3[k];
    out[b * 16 + i * 4 + i] = d;   // diagonal
  }
}

// ---------------------------------------------------------------------------
// Host launcher
// ---------------------------------------------------------------------------
extern "C" void kernel_launch(void* const* d_in, const int* in_sizes, int n_in,
                              void* d_out, int out_size, void* d_ws, size_t ws_size,
                              hipStream_t stream) {
  (void)in_sizes; (void)n_in; (void)out_size; (void)ws_size;

  const float*      x       = (const float*)d_in[0];
  const long long*  eidx    = (const long long*)d_in[1];
  const float*      eattr   = (const float*)d_in[2];
  const float*      enc_w1  = (const float*)d_in[5];
  const float*      enc_b1  = (const float*)d_in[6];
  const float*      enc_w2  = (const float*)d_in[7];
  const float*      enc_b2  = (const float*)d_in[8];
  const float*      cw10    = (const float*)d_in[9];
  const float*      cb10    = (const float*)d_in[10];
  const float*      cw11    = (const float*)d_in[11];
  const float*      cb11    = (const float*)d_in[12];
  const float*      cw12    = (const float*)d_in[13];
  const float*      cb12    = (const float*)d_in[14];
  const float*      cw20    = (const float*)d_in[15];
  const float*      cb20    = (const float*)d_in[16];
  const float*      cw21    = (const float*)d_in[17];
  const float*      cb21    = (const float*)d_in[18];
  const float*      cw22    = (const float*)d_in[19];
  const float*      cb22    = (const float*)d_in[20];
  const float*      dur_w1  = (const float*)d_in[21];
  const float*      dur_b1  = (const float*)d_in[22];
  const float*      dur_w2  = (const float*)d_in[23];
  const float*      dur_b2  = (const float*)d_in[24];
  const float*      dr_w1   = (const float*)d_in[25];
  const float*      dr_b1   = (const float*)d_in[26];
  const float*      dr_w2   = (const float*)d_in[27];
  const float*      dr_b2   = (const float*)d_in[28];
  const float*      dr_w3   = (const float*)d_in[29];
  const float*      dr_b3   = (const float*)d_in[30];

  const long long* srcIdx = eidx;
  const long long* dstIdx = eidx + NE;

  char* ws = (char*)d_ws;
  _Float16* wfrag = (_Float16*)(ws + OFF_WFRAG);
  float*    offs  = (float*)(ws + OFF_OFFS);
  _Float16* Hbuf  = (_Float16*)(ws + OFF_H);
  _Float16* T1    = (_Float16*)(ws + OFF_T1);
  _Float16* T2    = (_Float16*)(ws + OFF_T2);
  _Float16* Xn    = (_Float16*)(ws + OFF_XNODE);
  float*    agg   = (float*)(ws + OFF_AGG);
  _Float16* Xe    = (_Float16*)(ws + OFF_XEDGE);
  _Float16* MA    = (_Float16*)(ws + OFF_MA);
  _Float16* MB    = (_Float16*)(ws + OFF_MB);
  float*    out   = (float*)d_out;

  // --- 1) repack all GEMM weights into WMMA fragment layout -----------------
  struct WSpec { const float* w; int kreal; int kt; size_t foff; };
  WSpec wl[31];
  int nw = 0;
  wl[nw++] = {enc_w2, 64, 2, FR_ENC2};
  for (int l = 0; l < 5; ++l) {
    size_t base = FR_LAYER(l);
    wl[nw++] = {cw10 + (size_t)l * 72 * 64,  72, 3, base + 0};
    wl[nw++] = {cw11 + (size_t)l * 64 * 64,  64, 2, base + 6144};
    wl[nw++] = {cw12 + (size_t)l * 64 * 64,  64, 2, base + 10240};
    wl[nw++] = {cw20 + (size_t)l * 128 * 64, 128, 4, base + 14336};
    wl[nw++] = {cw21 + (size_t)l * 64 * 64,  64, 2, base + 22528};
    wl[nw++] = {cw22 + (size_t)l * 64 * 64,  64, 2, base + 26624};
  }
  for (int w = 0; w < nw; ++w) {
    int total = wl[w].kt * 2048;
    fl_wconv<<<(total + 255) / 256, 256, 0, stream>>>(
        wl[w].w, wfrag + wl[w].foff, wl[w].kreal, wl[w].kt);
  }

  // --- 2) per-i root offsets; zero `de` output ------------------------------
  fl_offsets<<<4, 256, 0, stream>>>(x, offs);
  fl_zero_f32<<<4, 256, 0, stream>>>(out + GB * 16, GB * 4);

  const int gE = NE / (128 * 4);  // 128 blocks: 8 waves x (4 tiles x 16 rows)
  const int gN = NTOT / 128;      // 72 blocks: 8 waves x (1 tile x 16 rows)

  // --- 3) the four Floquet sweeps ------------------------------------------
  for (int i = 0; i < 4; ++i) {
    const float* offs_i = offs + (size_t)i * GB;

    // encoder
    fl_encode<<<(NTOT * 64 + 255) / 256, 256, 0, stream>>>(x, offs_i, i, enc_w1, enc_b1, T1);
    fl_gemm<2, 1, false, false><<<gN, 256, 0, stream>>>(T1, 64, wfrag + FR_ENC2, enc_b2,
                                                        Hbuf, nullptr, nullptr);

    // 5 message-passing layers
    for (int l = 0; l < 5; ++l) {
      size_t fb = FR_LAYER(l);
      fl_gather_edge<<<(NE * 96 + 255) / 256, 256, 0, stream>>>(Hbuf, eattr, srcIdx, Xe);
      fl_gemm<3, 4, true, false><<<gE, 256, 0, stream>>>(Xe, 96, wfrag + fb + 0,
                                                         cb10 + l * 64, MA, nullptr, nullptr);
      fl_gemm<2, 4, true, false><<<gE, 256, 0, stream>>>(MA, 64, wfrag + fb + 6144,
                                                         cb11 + l * 64, MB, nullptr, nullptr);
      fl_zero_f32<<<(NTOT * 64 + 255) / 256, 256, 0, stream>>>(agg, NTOT * 64);
      fl_gemm<2, 4, false, true><<<gE, 256, 0, stream>>>(MB, 64, wfrag + fb + 10240,
                                                         cb12 + l * 64, nullptr, agg, dstIdx);
      fl_xnode<<<(NTOT * 128 + 255) / 256, 256, 0, stream>>>(Hbuf, agg, Xn);
      fl_gemm<4, 1, true, false><<<gN, 256, 0, stream>>>(Xn, 128, wfrag + fb + 14336,
                                                         cb20 + l * 64, T1, nullptr, nullptr);
      fl_gemm<2, 1, true, false><<<gN, 256, 0, stream>>>(T1, 64, wfrag + fb + 22528,
                                                         cb21 + l * 64, T2, nullptr, nullptr);
      fl_gemm<2, 1, false, false><<<gN, 256, 0, stream>>>(T2, 64, wfrag + fb + 26624,
                                                          cb22 + l * 64, Hbuf, nullptr, nullptr);
    }

    // collapse + dur MLP -> matrix row/col i ; decoder -> diagonal
    fl_dur<<<GB * 4, 64, 0, stream>>>(Hbuf, dur_w1, dur_b1, dur_w2, dur_b2, out, i);
    fl_dec<<<GB, 256, 0, stream>>>(Hbuf, offs_i, i, dr_w1, dr_b1, dr_w2, dr_b2,
                                   dr_w3, dr_b3, out);
  }
}
